// Model_8057358647773
// MI455X (gfx1250) — matrix operand order repaired and checked by reading, not compiled, not run on hardware
//
#include <hip/hip_runtime.h>
#include <hip/hip_bf16.h>
#include <stdint.h>

// Problem constants (match reference)
#define DDIM  512
#define LSEQ  200
#define NBAT  32
#define NHEAD 8
#define HDIM  64
#define NTOK  (NBAT * LSEQ)     // 6400 tokens
#define LN_EPS 1e-8f
#define KVSTRIDE (HDIM + 4)     // 68 bf16 = 136 B: 8B-aligned rows, conflict-free

typedef __attribute__((ext_vector_type(16))) __bf16 v16bf;
typedef __attribute__((ext_vector_type(8)))  __bf16 v8bf;
typedef __attribute__((ext_vector_type(8)))  float  v8f;

// Epilogue flags for the fused GEMM (compile-time template parameter)
#define EPI_RELU    1
#define EPI_RESID   2
#define EPI_MASK    4
#define EPI_F32OUT  8
#define EPI_BF16OUT 16

// ---------------------------------------------------------------------------
// Wave32 reductions
// ---------------------------------------------------------------------------
__device__ inline float wave_sum(float v) {
#pragma unroll
    for (int o = 16; o > 0; o >>= 1) v += __shfl_xor(v, o, 32);
    return v;
}
__device__ inline float wave_max(float v) {
#pragma unroll
    for (int o = 16; o > 0; o >>= 1) v = fmaxf(v, __shfl_xor(v, o, 32));
    return v;
}

// ---------------------------------------------------------------------------
// CDNA5 async global->LDS copy (8 bytes per lane), tracked by ASYNCcnt
// ---------------------------------------------------------------------------
__device__ inline void async_copy_b64(uint32_t lds_byte_addr, const void* gsrc) {
    asm volatile("global_load_async_to_lds_b64 %0, %1, off"
                 :: "v"(lds_byte_addr), "v"(gsrc) : "memory");
}
__device__ inline void wait_asynccnt0() {
    asm volatile("s_wait_asynccnt 0x0" ::: "memory");
}

// ---------------------------------------------------------------------------
// Weight convert + transpose: Wt[n*512+k] = bf16(W[k*512+n]); LDS-tiled.
// One launch per 512x512 matrix, 256 blocks (16x16 tiles of 32x32).
// ---------------------------------------------------------------------------
__global__ __launch_bounds__(256)
void wconv_kernel(const float* __restrict__ W, __bf16* __restrict__ Wt) {
    __shared__ float t[32][33];
    const int tk0 = (blockIdx.x & 15) * 32;
    const int tn0 = (blockIdx.x >> 4) * 32;
    const int c  = threadIdx.x & 31;
    const int r0 = threadIdx.x >> 5;   // 0..7
#pragma unroll
    for (int rr = 0; rr < 32; rr += 8)
        t[r0 + rr][c] = W[(size_t)(tk0 + r0 + rr) * DDIM + tn0 + c];
    __syncthreads();
#pragma unroll
    for (int rr = 0; rr < 32; rr += 8)
        Wt[(size_t)(tn0 + r0 + rr) * DDIM + tk0 + c] = (__bf16)t[c][r0 + rr];
}

// ---------------------------------------------------------------------------
// q = x * valid ; writes f32 + bf16 copies
// ---------------------------------------------------------------------------
__global__ void mask_mul_kernel(const float* __restrict__ x,
                                const uint8_t* __restrict__ tmask,
                                float* __restrict__ qf, __bf16* __restrict__ qb,
                                int n) {
    int i = blockIdx.x * blockDim.x + threadIdx.x;
    if (i < n) {
        int tok = i >> 9;                 // /DDIM
        float v = tmask[tok] ? 0.0f : x[i];
        qf[i] = v;
        qb[i] = (__bf16)v;
    }
}

// ---------------------------------------------------------------------------
// LayerNorm over D=512, one wave per token row; optional f32 / bf16 outputs
// ---------------------------------------------------------------------------
__global__ __launch_bounds__(256)
void layernorm512(const float* __restrict__ in, float* __restrict__ outF,
                  __bf16* __restrict__ outB,
                  const float* __restrict__ g, const float* __restrict__ b,
                  int rows) {
    const int lane = threadIdx.x & 31;
    const int wid  = threadIdx.x >> 5;
    const int row  = blockIdx.x * 8 + wid;
    if (row >= rows) return;

    const float* p = in + (size_t)row * DDIM;
    float v[16];
    float s = 0.0f;
#pragma unroll
    for (int j = 0; j < 16; ++j) { v[j] = p[lane + 32 * j]; s += v[j]; }
    s = wave_sum(s);
    const float mean = s * (1.0f / DDIM);

    float s2 = 0.0f;
#pragma unroll
    for (int j = 0; j < 16; ++j) { float d = v[j] - mean; s2 += d * d; }
    s2 = wave_sum(s2);
    const float rstd = rsqrtf(s2 * (1.0f / DDIM) + LN_EPS);

#pragma unroll
    for (int j = 0; j < 16; ++j) {
        int c = lane + 32 * j;
        float y = (v[j] - mean) * rstd * g[c] + b[c];
        if (outF) outF[(size_t)row * DDIM + c] = y;
        if (outB) outB[(size_t)row * DDIM + c] = (__bf16)y;
    }
}

// ---------------------------------------------------------------------------
// C[M,512] = A[M,512] @ W[512,512], A bf16 row-major, Wt bf16 TRANSPOSED
// ([N,K] row-major) so both fragments are contiguous 16/32-byte loads.
// Per-wave tile: 64(M) x 32(N) = 8 WMMAs per 32-wide K-step, A reused x2.
// EPI is compile-time: branch-free epilogues.
// ---------------------------------------------------------------------------
template <int EPI>
__global__ __launch_bounds__(256)
void gemm512(const __bf16* __restrict__ A, const __bf16* __restrict__ Wt,
             const float* __restrict__ bias, const float* __restrict__ resid,
             const uint8_t* __restrict__ tmask,
             float* __restrict__ Cf, __bf16* __restrict__ Cb, int M) {
    const int lane = threadIdx.x & 31;
    const int wid  = threadIdx.x >> 5;
    const int gw   = blockIdx.x * 8 + wid;
    const int ntiles = DDIM / 32;          // 16 N-tiles of 32
    const int m0 = (gw / ntiles) * 64;
    const int n0 = (gw % ntiles) * 32;
    if (m0 >= M) return;

    // Fragment-lane geometry; contiguous K order (k = 16*(lane/16) + s), same for A and B
    const int arow   = lane & 15;               // A: M = arow
    const int ka_off = (lane & 16) ? 16 : 0;    // A: lanes16-31 -> K 16..31
    const int ncol   = lane & 15;               // B/C: N = lane&15 within tile
    const int kb_off = (lane & 16) ? 16 : 0;    // B: lanes16-31 -> K 16..31

    v8f acc[4][2];
#pragma unroll
    for (int s = 0; s < 4; ++s) { acc[s][0] = v8f{}; acc[s][1] = v8f{}; }

    for (int k0 = 0; k0 < DDIM; k0 += 32) {
        // --- B fragments: Wt row = output column, contiguous 16 bf16
        v16bf b0 = *(const v16bf*)(Wt + (size_t)(n0 + ncol)      * DDIM + k0 + kb_off);
        v16bf b1 = *(const v16bf*)(Wt + (size_t)(n0 + 16 + ncol) * DDIM + k0 + kb_off);
        // --- 4 A fragments (16x32 each): two contiguous 8-element halves
#pragma unroll
        for (int s = 0; s < 4; ++s) {
            const __bf16* ap = A + (size_t)(m0 + s * 16 + arow) * DDIM + k0 + ka_off;
            v8bf lo = *(const v8bf*)(ap);        // K +0..7  (or +16..23)
            v8bf hi = *(const v8bf*)(ap + 8);    // K +8..15 (or +24..31)
            v16bf af;
#pragma unroll
            for (int j = 0; j < 8; ++j) { af[j] = lo[j]; af[j + 8] = hi[j]; }
            acc[s][0] = __builtin_amdgcn_wmma_f32_16x16x32_bf16(
                false, af, false, b0, (short)0, acc[s][0], false, false);
            acc[s][1] = __builtin_amdgcn_wmma_f32_16x16x32_bf16(
                false, af, false, b1, (short)0, acc[s][1], false, false);
        }
    }

    // --- epilogue: bias (+relu) (+residual) (+mask), f32 and/or bf16 out
#pragma unroll
    for (int t = 0; t < 2; ++t) {
        const int col  = n0 + t * 16 + ncol;
        const float bval = bias[col];
#pragma unroll
        for (int s = 0; s < 4; ++s) {
            const int mb = m0 + s * 16 + ((lane & 16) ? 8 : 0);
#pragma unroll
            for (int r = 0; r < 8; ++r) {
                const int row = mb + r;
                float v = acc[s][t][r] + bval;
                if constexpr (EPI & EPI_RELU)  v = fmaxf(v, 0.0f);
                if constexpr (EPI & EPI_RESID) v += resid[(size_t)row * DDIM + col];
                if constexpr (EPI & EPI_MASK)  v = tmask[row] ? 0.0f : v;
                if constexpr (EPI & EPI_F32OUT)  Cf[(size_t)row * DDIM + col] = v;
                if constexpr (EPI & EPI_BF16OUT) Cb[(size_t)row * DDIM + col] = (__bf16)v;
            }
        }
    }
}

// ---------------------------------------------------------------------------
// Causal attention per (batch, head); all-bf16 I/O. K/V head-slices staged
// into LDS via CDNA5 async global->LDS copies (ASYNCcnt). One wave per query
// row. Ctx may alias Q (row's q consumed before its ctx is written).
// ---------------------------------------------------------------------------
__global__ __launch_bounds__(256)
void attention(const __bf16* __restrict__ Q, const __bf16* __restrict__ K,
               const __bf16* __restrict__ V, __bf16* __restrict__ Ctx) {
    __shared__ __bf16 ksm[LSEQ][KVSTRIDE];
    __shared__ __bf16 vsm[LSEQ][KVSTRIDE];
    __shared__ float  qsm[8][HDIM];
    __shared__ float  ssm[8][LSEQ];

    const int b = blockIdx.x >> 3;
    const int h = blockIdx.x & 7;
    const int lane = threadIdx.x & 31;
    const int wid  = threadIdx.x >> 5;
    const size_t base = ((size_t)b * LSEQ) * DDIM + h * HDIM;

    // Async-copy K/V head slices: 8B (4 bf16) per lane per op.
    // chunks per matrix: LSEQ * (HDIM/4) = 3200
    for (int idx = threadIdx.x; idx < LSEQ * (HDIM / 4); idx += 256) {
        const int k = idx >> 4;            // row
        const int d = (idx & 15) * 4;      // dim start
        const __bf16* gk = K + base + (size_t)k * DDIM + d;
        const __bf16* gv = V + base + (size_t)k * DDIM + d;
        async_copy_b64((uint32_t)(uintptr_t)&ksm[k][d], gk);
        async_copy_b64((uint32_t)(uintptr_t)&vsm[k][d], gv);
    }
    wait_asynccnt0();
    __syncthreads();

    const float scale = 0.125f;  // 1/sqrt(64)
    for (int r = wid; r < LSEQ; r += 8) {
        qsm[wid][2 * lane]     = (float)Q[base + (size_t)r * DDIM + 2 * lane];
        qsm[wid][2 * lane + 1] = (float)Q[base + (size_t)r * DDIM + 2 * lane + 1];

        // scores for k <= r, lanes partition k
        float mloc = -3.0e38f;
        for (int k = lane; k <= r; k += 32) {
            float s = 0.0f;
#pragma unroll 8
            for (int d = 0; d < HDIM; ++d)
                s += qsm[wid][d] * (float)ksm[k][d];
            s *= scale;
            ssm[wid][k] = s;
            mloc = fmaxf(mloc, s);
        }
        const float rmax = wave_max(mloc);

        float sloc = 0.0f;
        for (int k = lane; k <= r; k += 32) {
            float e = __expf(ssm[wid][k] - rmax);
            ssm[wid][k] = e;
            sloc += e;
        }
        const float inv = 1.0f / wave_sum(sloc);

        // context: lanes partition d (2 dims each)
        float a0 = 0.0f, a1 = 0.0f;
        for (int k = 0; k <= r; ++k) {
            float p = ssm[wid][k];  // uniform address -> LDS broadcast
            a0 += p * (float)vsm[k][2 * lane];
            a1 += p * (float)vsm[k][2 * lane + 1];
        }
        Ctx[base + (size_t)r * DDIM + 2 * lane]     = (__bf16)(a0 * inv);
        Ctx[base + (size_t)r * DDIM + 2 * lane + 1] = (__bf16)(a1 * inv);
    }
}

// ---------------------------------------------------------------------------
// Orchestration
// ---------------------------------------------------------------------------
extern "C" void kernel_launch(void* const* d_in, const int* in_sizes, int n_in,
                              void* d_out, int out_size, void* d_ws, size_t ws_size,
                              hipStream_t stream) {
    const float*   x     = (const float*)d_in[0];
    const uint8_t* tmask = (const uint8_t*)d_in[1];   // bool [B,L]
    const float* ln1_g = (const float*)d_in[2];
    const float* ln1_b = (const float*)d_in[3];
    const float* Wi[6] = { (const float*)d_in[4],  (const float*)d_in[6],
                           (const float*)d_in[8],  (const float*)d_in[10],
                           (const float*)d_in[14], (const float*)d_in[16] };
    const float* bq = (const float*)d_in[5];
    const float* bk = (const float*)d_in[7];
    const float* bv = (const float*)d_in[9];
    const float* bo = (const float*)d_in[11];
    const float* ln2_g = (const float*)d_in[12];
    const float* ln2_b = (const float*)d_in[13];
    const float* b1 = (const float*)d_in[15];
    const float* b2 = (const float*)d_in[17];
    const float* lnf_g = (const float*)d_in[18];
    const float* lnf_b = (const float*)d_in[19];

    const size_t S  = (size_t)NTOK * DDIM;   // activation elements
    const size_t WS = (size_t)DDIM * DDIM;   // one weight matrix

    float*  qf  = (float*)d_ws;              // f32 stream
    float*  Qnf = qf + S;                    // f32 LN1 output (residual)
    __bf16* qb  = (__bf16*)(Qnf + S);        // bf16 stream
    __bf16* Qnb = qb + S;
    __bf16* Qt  = Qnb + S;                   // Q proj / ctx (in-place)
    __bf16* Kt  = Qt + S;                    // K proj / h1
    __bf16* Vt  = Kt + S;
    __bf16* Wt  = Vt + S;                    // 12 transposed bf16 weights

    const dim3 blk(256);
    const int gemmBlocks = (NTOK / 64) * (DDIM / 32) / 8;  // 200
    const int lnBlocks   = NTOK / 8;                        // 800

    // Pre-convert + transpose all weights to bf16 [N,K]
    for (int i = 0; i < 2; ++i)
        for (int j = 0; j < 6; ++j)
            wconv_kernel<<<256, blk, 0, stream>>>(Wi[j] + (size_t)i * WS,
                                                  Wt + (size_t)(i * 6 + j) * WS);

    mask_mul_kernel<<<(NTOK * DDIM + 255) / 256, blk, 0, stream>>>(
        x, tmask, qf, qb, NTOK * DDIM);

    for (int i = 0; i < 2; ++i) {
        const size_t oD = (size_t)i * DDIM;
        const __bf16* Wq = Wt + (size_t)(i * 6 + 0) * WS;
        const __bf16* Wk = Wt + (size_t)(i * 6 + 1) * WS;
        const __bf16* Wv = Wt + (size_t)(i * 6 + 2) * WS;
        const __bf16* Wo = Wt + (size_t)(i * 6 + 3) * WS;
        const __bf16* W1 = Wt + (size_t)(i * 6 + 4) * WS;
        const __bf16* W2 = Wt + (size_t)(i * 6 + 5) * WS;

        // Qn = LN1(q): f32 (residual for Wo) + bf16 (A of Wq)
        layernorm512<<<lnBlocks, blk, 0, stream>>>(qf, Qnf, Qnb,
                                                   ln1_g + oD, ln1_b + oD, NTOK);
        // Q/K/V projections (bf16 out only)
        gemm512<EPI_BF16OUT><<<gemmBlocks, blk, 0, stream>>>(
            Qnb, Wq, bq + oD, nullptr, nullptr, nullptr, Qt, NTOK);
        gemm512<EPI_BF16OUT><<<gemmBlocks, blk, 0, stream>>>(
            qb, Wk, bk + oD, nullptr, nullptr, nullptr, Kt, NTOK);
        gemm512<EPI_BF16OUT><<<gemmBlocks, blk, 0, stream>>>(
            qb, Wv, bv + oD, nullptr, nullptr, nullptr, Vt, NTOK);
        // causal attention; ctx overwrites Qt in place
        attention<<<NBAT * NHEAD, blk, 0, stream>>>(Qt, Kt, Vt, Qt);
        // q = ctx @ Wo + bo + Qn   (f32 out; feeds LN2)
        gemm512<EPI_RESID | EPI_F32OUT><<<gemmBlocks, blk, 0, stream>>>(
            Qt, Wo, bo + oD, Qnf, nullptr, qf, nullptr, NTOK);
        // q = LN2(q): f32 (residual for W2) + bf16 (A of W1), in place
        layernorm512<<<lnBlocks, blk, 0, stream>>>(qf, qf, qb,
                                                   ln2_g + oD, ln2_b + oD, NTOK);
        // h1 = relu(q @ W1 + b1) -> bf16 only (reuse Kt)
        gemm512<EPI_RELU | EPI_BF16OUT><<<gemmBlocks, blk, 0, stream>>>(
            qb, W1, b1 + oD, nullptr, nullptr, nullptr, Kt, NTOK);
        // q = (q + h1 @ W2 + b2) * valid   (f32 + bf16 for next block)
        gemm512<EPI_RESID | EPI_MASK | EPI_F32OUT | EPI_BF16OUT>
            <<<gemmBlocks, blk, 0, stream>>>(
            Kt, W2, b2 + oD, qf, tmask, qf, qb, NTOK);
    }
    // out = LN_f(q) in f32
    layernorm512<<<lnBlocks, blk, 0, stream>>>(qf, (float*)d_out, nullptr,
                                               lnf_g, lnf_b, NTOK);
}
